// BiGCN_77850577207792
// MI455X (gfx1250) — compile-verified, matrix operand order checked by reading, not run on hardware
//
#include <hip/hip_runtime.h>
#include <hip/hip_bf16.h>

#define N_NODES 100000
#define N_EDGES 1600000
#define N_GRAPHS 1000
#define IN_F 128
#define HID_F 64
#define OUT_F 4

typedef __attribute__((ext_vector_type(16))) __bf16 v16bf;
typedef __attribute__((ext_vector_type(8)))  float  v8f;

// ---------------------------------------------------------------------------
// Kernel 1: in-degree accumulation (into dinv buffer, pre-zeroed)
// ---------------------------------------------------------------------------
__global__ void deg_kernel(const int* __restrict__ dst, float* __restrict__ deg) {
    int e = blockIdx.x * blockDim.x + threadIdx.x;
    if (e < N_EDGES) {
        atomicAdd(&deg[dst[e]], 1.0f);
    }
}

// ---------------------------------------------------------------------------
// Kernel 2: dinv[n] = rsqrt(deg[n] + 1)   (self-loop folded in; deg+1 >= 1)
// ---------------------------------------------------------------------------
__global__ void dinv_kernel(float* __restrict__ dinv) {
    int n = blockIdx.x * blockDim.x + threadIdx.x;
    if (n < N_NODES) {
        dinv[n] = __frsqrt_rn(dinv[n] + 1.0f);
    }
}

// ---------------------------------------------------------------------------
// Kernel 3: h_td = x @ W_td, h_bu = x @ W_bu via v_wmma_f32_16x16x32_bf16.
// Block = 256 threads = 8 waves. Wave w: mat = w>>2 (td/bu), n-tile = w&3.
// Each block covers 16 rows of x (blockIdx.x), full 64-wide output, both mats.
// Fragment layouts per CDNA5 ISA 7.12.2 (16-bit A 16x32; 32-bit C/D 16x16).
// ---------------------------------------------------------------------------
__global__ void __launch_bounds__(256)
gemm_wmma_kernel(const float* __restrict__ x,
                 const float* __restrict__ Wtd,
                 const float* __restrict__ Wbu,
                 float* __restrict__ h_td,
                 float* __restrict__ h_bu) {
    const int l   = threadIdx.x & 31;       // lane in wave32
    const int w   = threadIdx.x >> 5;       // wave id 0..7
    const int mat = w >> 2;                 // 0 = td, 1 = bu
    const int nt  = w & 3;                  // n-tile (16 cols each)

    const float* __restrict__ Wm = mat ? Wbu : Wtd;
    float* __restrict__       Hm = mat ? h_bu : h_td;

    const int m_base = blockIdx.x * 16;
    const int lrow   = l & 15;              // M (for A) / N (for B) sub-index
    const int half   = l >> 4;              // 0: lanes 0-15, 1: lanes 16-31
    const int kb     = half * 8;            // K group base within 32-wide step
    const int row    = m_base + lrow;       // A row in x
    const int n      = nt * 16 + lrow;      // B/D column

    v8f acc = {};

    #pragma unroll
    for (int ks = 0; ks < IN_F; ks += 32) {
        v16bf a, b;
        // A fragment: lane holds row `row`, K = ks+kb+{0..7} and ks+kb+16+{0..7}
        const float* xr = x + (long)row * IN_F + ks + kb;
        #pragma unroll
        for (int i = 0; i < 8; ++i) {
            a[i]     = (__bf16)xr[i];
            a[i + 8] = (__bf16)xr[16 + i];
        }
        // B fragment: lane holds column n, same K striping (W is K x 64 row-major)
        const float* wr = Wm + (long)(ks + kb) * HID_F + n;
        #pragma unroll
        for (int i = 0; i < 8; ++i) {
            b[i]     = (__bf16)wr[i * HID_F];
            b[i + 8] = (__bf16)wr[(16 + i) * HID_F];
        }
        acc = __builtin_amdgcn_wmma_f32_16x16x32_bf16(
            /*neg_a=*/false, a, /*neg_b=*/false, b,
            /*c_mod=*/(short)0, acc, /*reuse_a=*/false, /*reuse_b=*/false);
    }

    // D layout: VGPR r -> M = r + 8*half, N = lane&15 (within tile)
    const int mrow = m_base + 8 * half;
    #pragma unroll
    for (int r = 0; r < 8; ++r) {
        Hm[(long)(mrow + r) * HID_F + n] = acc[r];
    }
}

// ---------------------------------------------------------------------------
// Kernel 4: init outputs with self-loop message + bias:
//   o = h * dinv^2 + b      (one thread per (node, feature))
// ---------------------------------------------------------------------------
__global__ void selfloop_kernel(const float* __restrict__ h_td,
                                const float* __restrict__ h_bu,
                                const float* __restrict__ dinv,
                                const float* __restrict__ b_td,
                                const float* __restrict__ b_bu,
                                float* __restrict__ o_td,
                                float* __restrict__ o_bu) {
    int t = blockIdx.x * blockDim.x + threadIdx.x;   // < N_NODES*64
    int n = t >> 6;
    int f = t & 63;
    float dv  = dinv[n];
    float nrm = dv * dv;
    o_td[t] = h_td[t] * nrm + b_td[f];
    o_bu[t] = h_bu[t] * nrm + b_bu[f];
}

// ---------------------------------------------------------------------------
// Kernel 5: edge scatter-add (the HBM-bound core). One thread per
// (edge, feature); both td and bu paths share src/dst/norm loads.
// ---------------------------------------------------------------------------
__global__ void edge_kernel(const int* __restrict__ src,
                            const int* __restrict__ dst,
                            const float* __restrict__ dinv,
                            const float* __restrict__ h_td,
                            const float* __restrict__ h_bu,
                            float* __restrict__ o_td,
                            float* __restrict__ o_bu) {
    long t = (long)blockIdx.x * blockDim.x + threadIdx.x;  // < N_EDGES*64
    int  e = (int)(t >> 6);
    int  f = (int)(t & 63);
    int  s = src[e];
    int  d = dst[e];
    float nrm = dinv[s] * dinv[d];
    atomicAdd(&o_td[(long)d * HID_F + f], h_td[(long)s * HID_F + f] * nrm);
    atomicAdd(&o_bu[(long)d * HID_F + f], h_bu[(long)s * HID_F + f] * nrm);
}

// ---------------------------------------------------------------------------
// Kernel 6: pooled feature sums. feat = [relu(td), td, relu(bu), bu] computed
// on the fly; one thread per (node, f in 0..63) -> 4 atomics into psum[g].
// ---------------------------------------------------------------------------
__global__ void pool_kernel(const float* __restrict__ o_td,
                            const float* __restrict__ o_bu,
                            const int* __restrict__ batch,
                            float* __restrict__ psum) {
    int t = blockIdx.x * blockDim.x + threadIdx.x;   // < N_NODES*64
    int n = t >> 6;
    int f = t & 63;
    int g = batch[n];
    float td = o_td[t];
    float bu = o_bu[t];
    float* pg = psum + (long)g * (4 * HID_F);
    atomicAdd(pg + f,                fmaxf(td, 0.0f));
    atomicAdd(pg + HID_F + f,        td);
    atomicAdd(pg + 2 * HID_F + f,    fmaxf(bu, 0.0f));
    atomicAdd(pg + 3 * HID_F + f,    bu);
}

// ---------------------------------------------------------------------------
// Kernel 7: per-graph node counts
// ---------------------------------------------------------------------------
__global__ void cnt_kernel(const int* __restrict__ batch, float* __restrict__ cnt) {
    int n = blockIdx.x * blockDim.x + threadIdx.x;
    if (n < N_NODES) {
        atomicAdd(&cnt[batch[n]], 1.0f);
    }
}

// ---------------------------------------------------------------------------
// Kernel 8: head: mean, fc (256x4), log_softmax. One thread per graph.
// ---------------------------------------------------------------------------
__global__ void head_kernel(const float* __restrict__ psum,
                            const float* __restrict__ cnt,
                            const float* __restrict__ fcW,
                            const float* __restrict__ fcb,
                            float* __restrict__ out) {
    int g = blockIdx.x * blockDim.x + threadIdx.x;
    if (g >= N_GRAPHS) return;
    float inv = 1.0f / fmaxf(cnt[g], 1.0f);
    float acc[OUT_F] = {fcb[0], fcb[1], fcb[2], fcb[3]};
    const float* p = psum + (long)g * (4 * HID_F);
    for (int c = 0; c < 4 * HID_F; ++c) {
        float v = p[c] * inv;
        #pragma unroll
        for (int o = 0; o < OUT_F; ++o) acc[o] += v * fcW[c * OUT_F + o];
    }
    float m = fmaxf(fmaxf(acc[0], acc[1]), fmaxf(acc[2], acc[3]));
    float s = 0.0f;
    #pragma unroll
    for (int o = 0; o < OUT_F; ++o) s += __expf(acc[o] - m);
    float lse = __logf(s);
    #pragma unroll
    for (int o = 0; o < OUT_F; ++o) out[(long)g * OUT_F + o] = acc[o] - m - lse;
}

// ---------------------------------------------------------------------------
// Launcher
// ---------------------------------------------------------------------------
extern "C" void kernel_launch(void* const* d_in, const int* in_sizes, int n_in,
                              void* d_out, int out_size, void* d_ws, size_t ws_size,
                              hipStream_t stream) {
    const float* x     = (const float*)d_in[0];   // [N, 128]
    const int*   ei    = (const int*)  d_in[1];   // [2, E]
    const int*   batch = (const int*)  d_in[2];   // [N]
    const float* W_td  = (const float*)d_in[3];   // [128, 64]
    const float* b_td  = (const float*)d_in[4];   // [64]
    const float* W_bu  = (const float*)d_in[5];   // [128, 64]
    const float* b_bu  = (const float*)d_in[6];   // [64]
    const float* fc_W  = (const float*)d_in[7];   // [256, 4]
    const float* fc_b  = (const float*)d_in[8];   // [4]
    float*       out   = (float*)d_out;           // [1000, 4]

    const int* src = ei;            // edge_index[0]
    const int* dst = ei + N_EDGES;  // edge_index[1]

    // Workspace layout (floats)
    float* ws   = (float*)d_ws;
    float* dinv = ws;                              // N      (deg -> dinv in place)
    float* h_td = dinv + N_NODES;                  // N*64
    float* h_bu = h_td + (long)N_NODES * HID_F;    // N*64
    float* o_td = h_bu + (long)N_NODES * HID_F;    // N*64
    float* o_bu = o_td + (long)N_NODES * HID_F;    // N*64
    float* psum = o_bu + (long)N_NODES * HID_F;    // G*256
    float* cnt  = psum + (long)N_GRAPHS * 4 * HID_F; // G

    // Zero accumulators (graph-capture-safe async memsets)
    hipMemsetAsync(dinv, 0, sizeof(float) * N_NODES, stream);
    hipMemsetAsync(psum, 0, sizeof(float) * N_GRAPHS * 4 * HID_F, stream);
    hipMemsetAsync(cnt,  0, sizeof(float) * N_GRAPHS, stream);

    // 1) degree + norm
    deg_kernel <<<(N_EDGES + 255) / 256, 256, 0, stream>>>(dst, dinv);
    dinv_kernel<<<(N_NODES + 255) / 256, 256, 0, stream>>>(dinv);

    // 2) dense transforms via WMMA (16 rows / block, both matrices)
    gemm_wmma_kernel<<<N_NODES / 16, 256, 0, stream>>>(x, W_td, W_bu, h_td, h_bu);

    // 3) self-loop message + bias (initializes o_td/o_bu)
    selfloop_kernel<<<(N_NODES * HID_F) / 256, 256, 0, stream>>>(
        h_td, h_bu, dinv, b_td, b_bu, o_td, o_bu);

    // 4) edge scatter-add (memory-bound core)
    edge_kernel<<<((long)N_EDGES * HID_F) / 256, 256, 0, stream>>>(
        src, dst, dinv, h_td, h_bu, o_td, o_bu);

    // 5) pooled sums + counts
    pool_kernel<<<(N_NODES * HID_F) / 256, 256, 0, stream>>>(o_td, o_bu, batch, psum);
    cnt_kernel <<<(N_NODES + 255) / 256, 256, 0, stream>>>(batch, cnt);

    // 6) mean + fc + log_softmax
    head_kernel<<<(N_GRAPHS + 255) / 256, 256, 0, stream>>>(psum, cnt, fc_W, fc_b, out);
}
